// KDAMixer_44203803410467
// MI455X (gfx1250) — matrix operand order repaired and checked by reading, not compiled, not run on hardware
//
#include <hip/hip_runtime.h>

// ---------------- CDNA5 WMMA types ----------------
typedef __attribute__((ext_vector_type(16))) _Float16 v16h;
typedef __attribute__((ext_vector_type(8)))  _Float16 h8;
typedef __attribute__((ext_vector_type(8)))  float    v8f;
typedef __attribute__((ext_vector_type(4)))  float    f4;

#define DEVI static __device__ __forceinline__

// ---------------- problem constants ----------------
constexpr int Bc  = 2;
constexpr int Tt  = 8192;
constexpr int Hh  = 16;
constexpr int Kc  = 32;
constexpr int Vc  = 64;
constexpr int BTc = 64;
constexpr int NTc = Tt / BTc;   // 128

// LDS pitches (units of the element type); multiples of 8 halves / 4 floats
// keep every b128 access 16B-aligned and dodge systematic bank conflicts.
constexpr int PK = 40;  // f16 pitch for 32-wide matrices
constexpr int PN = 72;  // f16 pitch for 64-wide matrices
constexpr int PF = 36;  // f32 pitch for 32-wide staging (16B-aligned quads)
constexpr int PS = 40;  // f32 pitch for transposed state rows (32 valid)

// ---------------- LDS layout (dynamic; ~117 KB of the WGP's 320 KB) -------
constexpr size_t OFF_GF   = 0;                          // g / gcum   64x36 f32
constexpr size_t OFF_QF   = OFF_GF   + 64 * PF * 4;
constexpr size_t OFF_KF   = OFF_QF   + 64 * PF * 4;
constexpr size_t OFF_ST   = OFF_KF   + 64 * PF * 4;     // ST[v][k]   64x40 f32
constexpr size_t OFF_BETA = OFF_ST   + 64 * PS * 4;
constexpr size_t OFF_GL   = OFF_BETA + 64 * 4;
constexpr size_t OFF_RQ   = OFF_GL   + 32 * 4;
constexpr size_t OFF_RK   = OFF_RQ   + 64 * 4;
constexpr size_t OFF_HQG  = OFF_RK   + 64 * 4;          // q*exp(g)   64x40 f16
constexpr size_t OFF_HKG  = OFF_HQG  + 64 * PK * 2;     // k*exp(g)
constexpr size_t OFF_HKD  = OFF_HKG  + 64 * PK * 2;     // k*exp(-g)
constexpr size_t OFF_HSTH = OFF_HKD  + 64 * PK * 2;     // f16 shadow of S^T
constexpr size_t OFF_HKGT = OFF_HSTH + 64 * PK * 2;     // (k*exp(g))^T 32x72
constexpr size_t OFF_HKLT = OFF_HKGT + 32 * PN * 2;     // k_dec^T      32x72
constexpr size_t OFF_HVT  = OFF_HKLT + 32 * PN * 2;     // v^T          64x72
constexpr size_t OFF_HVNT = OFF_HVT  + 64 * PN * 2;     // v_new^T      64x72
constexpr size_t OFF_HNW  = OFF_HVNT + 64 * PN * 2;     // -w           64x40
constexpr size_t OFF_HP   = OFF_HNW  + 64 * PK * 2;     // M powers / Aqk 64x72
constexpr size_t OFF_HPT  = OFF_HP   + 64 * PN * 2;     // transpose of hP
constexpr size_t OFF_HS   = OFF_HPT  + 64 * PN * 2;     // Neumann S / Afin
constexpr size_t SMEM_BYTES = OFF_HS + 64 * PN * 2;

// ---------------- CDNA5 async global->LDS copy (ISA 15.18.3, op 98) -------
// Per-lane: LDS[lds_off] = MEM[gaddr], 16 bytes, tracked by ASYNCcnt.
DEVI void async_copy_b128(unsigned lds_off, const float* gaddr) {
  asm volatile("global_load_async_to_lds_b128 %0, %1, off"
               :: "v"(lds_off), "v"(gaddr) : "memory");
}
DEVI void wait_async0() {
  asm volatile("s_wait_asynccnt 0" ::: "memory");
}

// ---------------- WMMA helpers ----------------
DEVI v8f wmma16(v16h a, v16h b, v8f c) {
  // D = A(16x32 f16) x B(32x16 f16) + C(16x16 f32)   -> v_wmma_f32_16x16x32_f16
  return __builtin_amdgcn_wmma_f32_16x16x32_f16(false, a, false, b,
                                                (short)0, c, false, false);
}

// A operand: 16x32 tile from row-major f16 buffer.
DEVI v16h load_a(const _Float16* buf, int pitch, int row0, int k0, int lane) {
  int r  = row0 + (lane & 15);
  int kk = k0 + ((lane >> 4) << 3);
  const _Float16* p = buf + r * pitch + kk;
  h8 lo = *(const h8*)p;
  h8 hi = *(const h8*)(p + 16);
  v16h a;
#pragma unroll
  for (int i = 0; i < 8; ++i) { a[i] = lo[i]; a[i + 8] = hi[i]; }
  return a;
}

// B operand: 32x16 tile; column n comes from row n of a transposed buffer.
DEVI v16h load_bT(const _Float16* bufT, int pitch, int k0, int n0, int lane) {
  int c  = n0 + (lane & 15);
  int kk = k0 + ((lane >> 4) << 3);
  const _Float16* p = bufT + c * pitch + kk;
  h8 lo = *(const h8*)p;
  h8 hi = *(const h8*)(p + 16);
  v16h b;
#pragma unroll
  for (int i = 0; i < 8; ++i) { b[i] = lo[i]; b[i + 8] = hi[i]; }
  return b;
}

// C accumulator init from f16 LDS (matches C/D VGPR layout).
DEVI v8f load_c(const _Float16* buf, int pitch, int row0, int n0, int lane) {
  int n  = n0 + (lane & 15);
  int m0 = row0 + ((lane >> 4) << 3);
  v8f c;
#pragma unroll
  for (int r = 0; r < 8; ++r) c[r] = (float)buf[(m0 + r) * pitch + n];
  return c;
}

// ---------------- fused chunked KDA scan ----------------
// grid = B*H workgroups; block = 128 threads = 4 waves (wave32).
// Each workgroup scans its 128 chunks serially, carrying S^T in LDS
// (f32 master for accumulation + f16 shadow as the WMMA B-operand).
__global__ __launch_bounds__(128, 1)
void kda_chunk_scan(const float* __restrict__ qg_, const float* __restrict__ kg_,
                    const float* __restrict__ vg_, const float* __restrict__ graw_,
                    const float* __restrict__ beta_, const float* __restrict__ Alog_,
                    const float* __restrict__ dtb_, float* __restrict__ out_) {
  extern __shared__ char smem[];
  float*    gF    = (float*)(smem + OFF_GF);
  float*    qF    = (float*)(smem + OFF_QF);
  float*    kF    = (float*)(smem + OFF_KF);
  float*    stT   = (float*)(smem + OFF_ST);
  float*    sBeta = (float*)(smem + OFF_BETA);
  float*    gLast = (float*)(smem + OFF_GL);
  float*    rqS   = (float*)(smem + OFF_RQ);
  float*    rkS   = (float*)(smem + OFF_RK);
  _Float16* hQg   = (_Float16*)(smem + OFF_HQG);
  _Float16* hKg   = (_Float16*)(smem + OFF_HKG);
  _Float16* hKd   = (_Float16*)(smem + OFF_HKD);
  _Float16* hSTh  = (_Float16*)(smem + OFF_HSTH);
  _Float16* hKgT  = (_Float16*)(smem + OFF_HKGT);
  _Float16* hKlT  = (_Float16*)(smem + OFF_HKLT);
  _Float16* hVT   = (_Float16*)(smem + OFF_HVT);
  _Float16* hVnT  = (_Float16*)(smem + OFF_HVNT);
  _Float16* hNw   = (_Float16*)(smem + OFF_HNW);
  _Float16* hP    = (_Float16*)(smem + OFF_HP);
  _Float16* hPT   = (_Float16*)(smem + OFF_HPT);
  _Float16* hS    = (_Float16*)(smem + OFF_HS);

  const int tid  = threadIdx.x;
  const int lane = tid & 31;
  const int wave = tid >> 5;
  const int row0 = wave * 16;
  const int nlo  = lane & 15;                 // column-within-tile
  const int m0   = row0 + ((lane >> 4) << 3); // accumulator row base (64-wide)
  const int bb   = blockIdx.x >> 4;   // H = 16
  const int hh   = blockIdx.x & 15;

  const float expA = __expf(Alog_[hh]);
  const float* dtb = dtb_ + hh * Kc;

  // zero the carried state (f32 master + f16 shadow)
  for (int i = tid; i < Vc * PS; i += 128) stT[i] = 0.f;
  for (int i = tid; i < Vc * PK; i += 128) hSTh[i] = (_Float16)0.f;
  __syncthreads();

  for (int c = 0; c < NTc; ++c) {
    const int t0 = c * BTc;

    // ---- stage q,k via CDNA5 async global->LDS DMA (no VGPR round trip) ----
    // 64 rows x 8 quads = 512 b128 transfers per buffer; 4 per lane.
#pragma unroll
    for (int e = 0; e < 4; ++e) {
      int idx = tid + e * 128;
      int t = idx >> 3, kq = (idx & 7) << 2;
      int gi = ((bb * Tt + t0 + t) * Hh + hh) * Kc + kq;
      unsigned qoff = (unsigned)(size_t)(qF + t * PF + kq);
      unsigned koff = (unsigned)(size_t)(kF + t * PF + kq);
      async_copy_b128(qoff, qg_ + gi);
      async_copy_b128(koff, kg_ + gi);
    }
    // ---- g gate (needs softplus transform -> VALU path) and v^T (f16) ----
    for (int e = 0; e < 16; ++e) {
      int idx = tid + e * 128;
      int t = idx >> 5, kk = idx & 31;
      float x  = graw_[((bb * Tt + t0 + t) * Hh + hh) * Kc + kk] + dtb[kk];
      float sp = (x > 15.f) ? x : log1pf(__expf(x));
      gF[t * PF + kk] = -expA * sp;                 // g <= 0 always
    }
    for (int e = 0; e < 32; ++e) {
      int idx = tid + e * 128;
      int t = idx >> 6, n = idx & 63;
      hVT[n * PN + t] =
          (_Float16)vg_[((bb * Tt + t0 + t) * Hh + hh) * Vc + n];
    }
    if (tid < 64) sBeta[tid] = beta_[(bb * Tt + t0 + tid) * Hh + hh];
    if (c + 1 < NTc) {  // hide next chunk's HBM latency behind this chunk's math
      int t = tid & 63;
      int gi = ((bb * Tt + t0 + BTc + t) * Hh + hh) * Kc;
      __builtin_prefetch(graw_ + gi, 0, 1);
      __builtin_prefetch(vg_ + ((bb * Tt + t0 + BTc + t) * Hh + hh) * Vc, 0, 1);
    }
    wait_async0();
    __syncthreads();

    // ---- gate cumsum (32 lanes, one K-column each) + q/k row l2-norms
    if (tid < 32) {
      float run = 0.f;
      for (int t = 0; t < BTc; ++t) { run += gF[t * PF + tid]; gF[t * PF + tid] = run; }
      gLast[tid] = run;
    } else if (tid >= 64) {
      int r = tid - 64;
      float sq = 0.f, sk = 0.f;
      for (int kk = 0; kk < Kc; ++kk) {
        float a = qF[r * PF + kk]; sq += a * a;
        float b = kF[r * PF + kk]; sk += b * b;
      }
      rqS[r] = 0.17677669529663687f / fmaxf(sqrtf(sq), 1e-6f);  // * K^-0.5
      rkS[r] = 1.f / fmaxf(sqrtf(sk), 1e-6f);
    }
    __syncthreads();

    // ---- gated f16 operands. decay factorizes on the lower triangle since
    // g<=0: exp(gc_i - gc_j) = exp(gc_i)*exp(-gc_j); exp(-gc) clamped f16-safe.
    for (int e = 0; e < 16; ++e) {
      int idx = tid + e * 128;
      int t = idx >> 5, kk = idx & 31;
      float gc  = gF[t * PF + kk];
      float eg  = __expf(gc);
      float egi = __expf(fminf(-gc, 11.f));
      float el  = __expf(gLast[kk] - gc);           // exponent <= 0
      float qv  = qF[t * PF + kk] * rqS[t];
      float kv  = kF[t * PF + kk] * rkS[t];
      hQg[t * PK + kk]  = (_Float16)(qv * eg);
      float kgv = kv * eg;
      hKg[t * PK + kk]  = (_Float16)kgv;
      hKgT[kk * PN + t] = (_Float16)kgv;
      hKd[t * PK + kk]  = (_Float16)(kv * egi);
      hKlT[kk * PN + t] = (_Float16)(kv * el);
    }
    __syncthreads();

    // Hoisted, unconditional per-lane LDS reads used by masked writes below
    // (keeps masks as pure register cndmasks -- no exec-mask gymnastics).
    f4 bA = *(const f4*)(sBeta + m0);
    f4 bB = *(const f4*)(sBeta + m0 + 4);
    float bRow[8];
#pragma unroll
    for (int r = 0; r < 4; ++r) { bRow[r] = bA[r]; bRow[4 + r] = bB[r]; }
    float bCol[4];
#pragma unroll
    for (int jt = 0; jt < 4; ++jt) bCol[jt] = sBeta[jt * 16 + nlo];

    // ---- Akk = kg @ kd^T  ->  M = -tril_strict(Akk)*beta_row,  S0 = I + M
    {
      v16h a = load_a(hKg, PK, row0, 0, lane);
      v8f acc[4];
#pragma unroll
      for (int jt = 0; jt < 4; ++jt) {
        v16h b = load_bT(hKd, PK, 0, jt * 16, lane);
        v8f z = {};
        acc[jt] = wmma16(a, b, z);
      }
#pragma unroll
      for (int jt = 0; jt < 4; ++jt) {
        int n = jt * 16 + nlo;
        h8 tv;
#pragma unroll
        for (int r = 0; r < 8; ++r) {
          int i = m0 + r;
          float mv = (i > n) ? (-bRow[r] * acc[jt][r]) : 0.f;
          _Float16 hv = (_Float16)mv;
          tv[r] = hv;
          hP[i * PN + n] = hv;
          hS[i * PN + n] = (_Float16)((i == n) ? 1.f : mv);
        }
        *(h8*)(hPT + n * PN + m0) = tv;   // contiguous b128 store
      }
    }
    __syncthreads();

    // ---- (I-M)^-1 via nilpotent Neumann product: prod_{j=0..5}(I + M^(2^j))
    for (int it = 0; it < 5; ++it) {
      v8f pn[4];
      {
        v16h a0 = load_a(hP, PN, row0, 0, lane);
        v16h a1 = load_a(hP, PN, row0, 32, lane);
#pragma unroll
        for (int jt = 0; jt < 4; ++jt) {
          v16h b0 = load_bT(hPT, PN, 0,  jt * 16, lane);
          v16h b1 = load_bT(hPT, PN, 32, jt * 16, lane);
          v8f z = {};
          v8f a2 = wmma16(a0, b0, z);
          pn[jt]  = wmma16(a1, b1, a2);               // P <- P @ P
        }
      }
      __syncthreads();
#pragma unroll
      for (int jt = 0; jt < 4; ++jt) {
        int n = jt * 16 + nlo;
        h8 tv;
#pragma unroll
        for (int r = 0; r < 8; ++r) {
          _Float16 hv = (_Float16)pn[jt][r];
          tv[r] = hv;
          hP[(m0 + r) * PN + n] = hv;
        }
        *(h8*)(hPT + n * PN + m0) = tv;
      }
      __syncthreads();
      v8f sn[4];
      {
        v16h a0 = load_a(hS, PN, row0, 0, lane);
        v16h a1 = load_a(hS, PN, row0, 32, lane);
#pragma unroll
        for (int jt = 0; jt < 4; ++jt) {
          v8f a2 = load_c(hS, PN, row0, jt * 16, lane); // S + S@Pnew
          v16h b0 = load_bT(hPT, PN, 0,  jt * 16, lane);
          v16h b1 = load_bT(hPT, PN, 32, jt * 16, lane);
          a2 = wmma16(a0, b0, a2);
          sn[jt] = wmma16(a1, b1, a2);
        }
      }
      __syncthreads();
      const bool last = (it == 4);
#pragma unroll
      for (int jt = 0; jt < 4; ++jt) {
        int n = jt * 16 + nlo;
        float cs = last ? bCol[jt] : 1.f;   // register select; fold Afin=T*beta
#pragma unroll
        for (int r = 0; r < 8; ++r)
          hS[(m0 + r) * PN + n] = (_Float16)(sn[jt][r] * cs);
      }
      __syncthreads();
    }
    // hS = Afin

    // ---- Aqk = qg @ kd^T (tril incl, reuse hP)  and  -w = -(Afin @ kg)
    {
      v16h aq = load_a(hQg, PK, row0, 0, lane);
      v8f aqk[4];
#pragma unroll
      for (int jt = 0; jt < 4; ++jt) {
        v16h b = load_bT(hKd, PK, 0, jt * 16, lane);
        v8f z = {};
        aqk[jt] = wmma16(aq, b, z);
      }
      v16h af0 = load_a(hS, PN, row0, 0, lane);
      v16h af1 = load_a(hS, PN, row0, 32, lane);
      v8f wv[2];
#pragma unroll
      for (int jt = 0; jt < 2; ++jt) {
        v16h b0 = load_bT(hKgT, PN, 0,  jt * 16, lane);
        v16h b1 = load_bT(hKgT, PN, 32, jt * 16, lane);
        v8f z = {};
        v8f a2 = wmma16(af0, b0, z);
        wv[jt]  = wmma16(af1, b1, a2);
      }
#pragma unroll
      for (int jt = 0; jt < 4; ++jt) {
        int n = jt * 16 + nlo;
#pragma unroll
        for (int r = 0; r < 8; ++r) {
          int i = m0 + r;
          hP[i * PN + n] = (_Float16)((i >= n) ? aqk[jt][r] : 0.f);
        }
      }
#pragma unroll
      for (int jt = 0; jt < 2; ++jt) {
        int n = jt * 16 + nlo;
#pragma unroll
        for (int r = 0; r < 8; ++r)
          hNw[(m0 + r) * PK + n] = (_Float16)(-wv[jt][r]);
      }
    }
    __syncthreads();

    // ---- v_new = Afin@v + (-w)@S   (store transposed, b128)
    {
      v16h af0 = load_a(hS, PN, row0, 0, lane);
      v16h af1 = load_a(hS, PN, row0, 32, lane);
      v16h aw  = load_a(hNw, PK, row0, 0, lane);
      v8f vn[4];
#pragma unroll
      for (int jt = 0; jt < 4; ++jt) {
        v16h b0 = load_bT(hVT, PN, 0,  jt * 16, lane);
        v16h b1 = load_bT(hVT, PN, 32, jt * 16, lane);
        v16h bs = load_bT(hSTh, PK, 0, jt * 16, lane);  // f16 shadow of S^T
        v8f z = {};
        v8f a2 = wmma16(af0, b0, z);
        a2 = wmma16(af1, b1, a2);
        vn[jt] = wmma16(aw, bs, a2);
      }
#pragma unroll
      for (int jt = 0; jt < 4; ++jt) {
        int n = jt * 16 + nlo;
        h8 tv;
#pragma unroll
        for (int r = 0; r < 8; ++r) tv[r] = (_Float16)vn[jt][r];
        *(h8*)(hVnT + n * PN + m0) = tv;
      }
    }
    __syncthreads();

    // ---- o = qg@S + Aqk@v_new  -> global (f32)
    {
      v16h aq  = load_a(hQg, PK, row0, 0, lane);
      v16h aa0 = load_a(hP, PN, row0, 0, lane);
      v16h aa1 = load_a(hP, PN, row0, 32, lane);
#pragma unroll
      for (int jt = 0; jt < 4; ++jt) {
        v16h bs = load_bT(hSTh, PK, 0, jt * 16, lane);
        v16h b0 = load_bT(hVnT, PN, 0,  jt * 16, lane);
        v16h b1 = load_bT(hVnT, PN, 32, jt * 16, lane);
        v8f z = {};
        v8f a2 = wmma16(aq, bs, z);
        a2 = wmma16(aa0, b0, a2);
        a2 = wmma16(aa1, b1, a2);
        int n = jt * 16 + nlo;
#pragma unroll
        for (int r = 0; r < 8; ++r)
          out_[((bb * Tt + t0 + m0 + r) * Hh + hh) * Vc + n] = a2[r];
      }
    }
    __syncthreads();

    // ---- state: S^T <- S^T * exp(g_last) + (klT @ v_new)^T  (8 tiles/4 waves)
    //      f32 master RMW (two float4) + refresh the f16 shadow (one b128).
    {
      int kt = wave >> 1;
      int km0 = kt * 16 + ((lane >> 4) << 3);       // k index base (0..31)
      f4 g0 = *(const f4*)(gLast + km0);
      f4 g1 = *(const f4*)(gLast + km0 + 4);
      float el[8];
#pragma unroll
      for (int r = 0; r < 4; ++r) { el[r] = __expf(g0[r]); el[4 + r] = __expf(g1[r]); }
      v16h a0 = load_a(hKlT, PN, kt * 16, 0, lane);
      v16h a1 = load_a(hKlT, PN, kt * 16, 32, lane);
#pragma unroll
      for (int p = 0; p < 2; ++p) {
        int ct = ((wave & 1) << 1) + p;
        v16h b0 = load_bT(hVnT, PN, 0,  ct * 16, lane);
        v16h b1 = load_bT(hVnT, PN, 32, ct * 16, lane);
        v8f z = {};
        v8f a2 = wmma16(a0, b0, z);
        a2 = wmma16(a1, b1, a2);
        int n = ct * 16 + nlo;                       // v index
        float* sp = stT + n * PS + km0;
        f4 s0 = *(const f4*)sp;
        f4 s1 = *(const f4*)(sp + 4);
        h8 tv;
#pragma unroll
        for (int r = 0; r < 4; ++r) {
          s0[r] = s0[r] * el[r]     + a2[r];
          s1[r] = s1[r] * el[4 + r] + a2[4 + r];
          tv[r]     = (_Float16)s0[r];
          tv[4 + r] = (_Float16)s1[r];
        }
        *(f4*)sp       = s0;
        *(f4*)(sp + 4) = s1;
        *(h8*)(hSTh + n * PK + km0) = tv;
      }
    }
    __syncthreads();
  }
}

extern "C" void kernel_launch(void* const* d_in, const int* in_sizes, int n_in,
                              void* d_out, int out_size, void* d_ws, size_t ws_size,
                              hipStream_t stream) {
  (void)in_sizes; (void)n_in; (void)d_ws; (void)ws_size; (void)out_size;
  const float* q       = (const float*)d_in[0];
  const float* k       = (const float*)d_in[1];
  const float* v       = (const float*)d_in[2];
  const float* g_raw   = (const float*)d_in[3];
  const float* beta    = (const float*)d_in[4];
  const float* A_log   = (const float*)d_in[5];
  const float* dt_bias = (const float*)d_in[6];

  // ~117 KB dynamic LDS per workgroup (WGP has 320 KB); opt in explicitly.
  (void)hipFuncSetAttribute(reinterpret_cast<const void*>(kda_chunk_scan),
                            hipFuncAttributeMaxDynamicSharedMemorySize,
                            (int)SMEM_BYTES);
  kda_chunk_scan<<<dim3(Bc * Hh), dim3(128), SMEM_BYTES, stream>>>(
      q, k, v, g_raw, beta, A_log, dt_bias, (float*)d_out);
}
